// Recurrent_Net_29652454211937
// MI455X (gfx1250) — matrix operand order repaired and checked by reading, not compiled, not run on hardware
//
#include <hip/hip_runtime.h>

// ---- problem constants (match reference) ----
#define BATCH 64
#define SEQT  256
#define HID   512       // D == H == 512
#define NLAY  32
#define FCOUT 256

typedef _Float16 v16h __attribute__((ext_vector_type(16)));
typedef _Float16 v8h  __attribute__((ext_vector_type(8)));
typedef float    v8f  __attribute__((ext_vector_type(8)));

// ---------------------------------------------------------------------------
// WMMA fragment loaders (layouts per CDNA5 ISA 7.12.2, 16-bit A 16x32 / B 32x16)
// A (MxK=16x32): lane L -> M = L%16 ; kBase = (L/16)*8 ; elems 0..7 = K kBase..kBase+7,
//                elems 8..15 = K kBase+16..kBase+23.  Two contiguous 16B chunks.
// B (KxN=32x16): lane L -> N = L%16 ; kBase = (L/16)*8 ; elems 0..7 = K kBase..kBase+7, elems 8..15 = K kBase+16..kBase+23.
//                B[k][n] = W[n][k] (row-major W) -> K contiguous in memory.
// ---------------------------------------------------------------------------
__device__ __forceinline__ v16h load_a_frag(const _Float16* __restrict__ base,
                                            int stride, int mrow0, int k0, int lane) {
  int m  = lane & 15;
  int kb = (lane >> 4) * 8;
  const _Float16* p = base + (size_t)(mrow0 + m) * stride + k0 + kb;
  union { v16h v; v8h h[2]; } u;
  u.h[0] = *(const v8h*)(p);
  u.h[1] = *(const v8h*)(p + 16);
  return u.v;
}

__device__ __forceinline__ v16h load_a_frag_lds(const _Float16* base,
                                                int mrow0, int k0, int lane) {
  int m  = lane & 15;
  int kb = (lane >> 4) * 8;
  const _Float16* p = base + (mrow0 + m) * HID + k0 + kb;
  union { v16h v; v8h h[2]; } u;
  u.h[0] = *(const v8h*)(p);
  u.h[1] = *(const v8h*)(p + 16);
  return u.v;
}

__device__ __forceinline__ v16h load_b_frag(const _Float16* __restrict__ base,
                                            int stride, int ncol0, int k0, int lane) {
  int n  = lane & 15;
  int kb = (lane >> 4) * 8;
  const _Float16* p = base + (size_t)(ncol0 + n) * stride + k0 + kb;
  union { v16h v; v8h h[2]; } u;
  u.h[0] = *(const v8h*)(p);
  u.h[1] = *(const v8h*)(p + 16);
  return u.v;
}

#define WMMA_F16(a, b, c) \
  __builtin_amdgcn_wmma_f32_16x16x32_f16(false, (a), false, (b), (short)0, (c), false, false)

// ---------------------------------------------------------------------------
// One RNN layer: persistent single workgroup, 32 waves (wave32), h in LDS.
// Wave w owns hidden columns [16w, 16w+16), loops 4 batch M-tiles of 16.
//   seq_in  : [T][B][512] f16   seq_out : [T][B][512] f16
// K loops: unroll(2) + branchless wrapped prefetch of the B (weight) fragment.
// The two prefetch buffers alternate registers across the unrolled pair, so no
// v_mov rotation, no branch diamond, no WMMA WAR-hazard nops; full unroll is
// avoided because it spilled WMMA operands to scratch (round-1 result).
// ---------------------------------------------------------------------------
__global__ __launch_bounds__(1024) void rnn_layer_kernel(
    const _Float16* __restrict__ wih,   // [H][D]  f16, this layer
    const _Float16* __restrict__ whh,   // [H][H]  f16
    const float*    __restrict__ bih,   // [H]
    const float*    __restrict__ bhh,   // [H]
    const _Float16* __restrict__ seq_in,
    _Float16*       __restrict__ seq_out)
{
  __shared__ _Float16 hprev[BATCH * HID];   // 64 KB of the 320 KB WGP LDS

  const int tid   = threadIdx.x;
  const int lane  = tid & 31;
  const int wave  = tid >> 5;          // 0..31
  const int ncol0 = wave * 16;

  // h0 = 0
  for (int i = tid; i < BATCH * HID; i += 1024) hprev[i] = (_Float16)0.0f;
  __syncthreads();

  const int   col     = ncol0 + (lane & 15);
  const int   rowHalf = (lane >> 4) * 8;
  const float bias    = bih[col] + bhh[col];

  for (int t = 0; t < SEQT; ++t) {
    const _Float16* xt = seq_in + (size_t)t * BATCH * HID;

    v8f acc[4];
    #pragma unroll
    for (int m = 0; m < 4; ++m)
      #pragma unroll
      for (int r = 0; r < 8; ++r) acc[m][r] = bias;

    // ---- input contribution: X[t] (B x D) * Wih^T ----
    {
      v16h bf = load_b_frag(wih, HID, ncol0, 0, lane);
      #pragma unroll 2
      for (int k0 = 0; k0 < HID; k0 += 32) {
        // branchless prefetch; wraps to k=0 on the last iteration (harmless)
        v16h bf_next = load_b_frag(wih, HID, ncol0, (k0 + 32) & (HID - 1), lane);
        #pragma unroll
        for (int m = 0; m < 4; ++m) {
          v16h af = load_a_frag(xt, HID, m * 16, k0, lane);
          acc[m] = WMMA_F16(af, bf, acc[m]);
        }
        bf = bf_next;
      }
    }
    // ---- recurrent contribution: H[t-1] (B x H) * Whh^T  (A from LDS) ----
    {
      v16h bf = load_b_frag(whh, HID, ncol0, 0, lane);
      #pragma unroll 2
      for (int k0 = 0; k0 < HID; k0 += 32) {
        v16h bf_next = load_b_frag(whh, HID, ncol0, (k0 + 32) & (HID - 1), lane);
        #pragma unroll
        for (int m = 0; m < 4; ++m) {
          v16h af = load_a_frag_lds(hprev, m * 16, k0, lane);
          acc[m] = WMMA_F16(af, bf, acc[m]);
        }
        bf = bf_next;
      }
    }

    __syncthreads();   // all waves finished reading hprev

    _Float16* yt = seq_out + (size_t)t * BATCH * HID;
    #pragma unroll
    for (int m = 0; m < 4; ++m) {
      #pragma unroll
      for (int r = 0; r < 8; ++r) {
        int row = m * 16 + rowHalf + r;
        _Float16 hv = (_Float16)tanhf(acc[m][r]);
        hprev[row * HID + col]      = hv;
        yt[(size_t)row * HID + col] = hv;
      }
    }
    __syncthreads();   // new h visible to all waves
  }
}

// ---------------------------------------------------------------------------
// FC head: out[64,256] = relu(h_last[64,512] @ Wfc^T + b_fc), 16 waves.
// ---------------------------------------------------------------------------
__global__ __launch_bounds__(512) void fc_kernel(
    const _Float16* __restrict__ hlast,  // [B][H] f16 (= seq_out[T-1] of last layer)
    const _Float16* __restrict__ wfc,    // [OUT][H] f16
    const float*    __restrict__ bfc,    // [OUT]
    float*          __restrict__ out)    // [B][OUT]
{
  const int tid   = threadIdx.x;
  const int lane  = tid & 31;
  const int wave  = tid >> 5;          // 0..15
  const int ncol0 = wave * 16;
  const int col   = ncol0 + (lane & 15);
  const int rowHalf = (lane >> 4) * 8;
  const float bias = bfc[col];

  v8f acc[4];
  #pragma unroll
  for (int m = 0; m < 4; ++m)
    #pragma unroll
    for (int r = 0; r < 8; ++r) acc[m][r] = bias;

  {
    v16h bf = load_b_frag(wfc, HID, ncol0, 0, lane);
    #pragma unroll 2
    for (int k0 = 0; k0 < HID; k0 += 32) {
      v16h bf_next = load_b_frag(wfc, HID, ncol0, (k0 + 32) & (HID - 1), lane);
      #pragma unroll
      for (int m = 0; m < 4; ++m) {
        v16h af = load_a_frag(hlast, HID, m * 16, k0, lane);
        acc[m] = WMMA_F16(af, bf, acc[m]);
      }
      bf = bf_next;
    }
  }

  #pragma unroll
  for (int m = 0; m < 4; ++m)
    #pragma unroll
    for (int r = 0; r < 8; ++r) {
      int row = m * 16 + rowHalf + r;
      out[(size_t)row * FCOUT + col] = fmaxf(acc[m][r], 0.0f);
    }
}

// ---------------------------------------------------------------------------
// One-shot conversions: fp32 -> f16 weights; x [B][T][D] -> time-major f16.
// ---------------------------------------------------------------------------
__global__ void cvt_f32_to_f16(const float* __restrict__ in,
                               _Float16* __restrict__ out, int n) {
  int i = blockIdx.x * blockDim.x + threadIdx.x;
  if (i < n) out[i] = (_Float16)in[i];
}

__global__ void transpose_x_kernel(const float* __restrict__ x,   // [B][T][D]
                                   _Float16* __restrict__ xt) {   // [T][B][D]
  int i = blockIdx.x * blockDim.x + threadIdx.x;        // enumerates (t,b,d)
  if (i >= BATCH * SEQT * HID) return;
  int d = i & (HID - 1);
  int b = (i >> 9) & (BATCH - 1);
  int t = i >> 15;
  xt[i] = (_Float16)x[(size_t)b * SEQT * HID + (size_t)t * HID + d];
}

// ---------------------------------------------------------------------------
extern "C" void kernel_launch(void* const* d_in, const int* in_sizes, int n_in,
                              void* d_out, int out_size, void* d_ws, size_t ws_size,
                              hipStream_t stream) {
  const float* x    = (const float*)d_in[0];   // [B][T][D]
  const float* W_ih = (const float*)d_in[1];   // [L][H][D]
  const float* W_hh = (const float*)d_in[2];   // [L][H][H]
  const float* b_ih = (const float*)d_in[3];   // [L][H]
  const float* b_hh = (const float*)d_in[4];   // [L][H]
  const float* W_fc = (const float*)d_in[5];   // [OUT][H]
  const float* b_fc = (const float*)d_in[6];   // [OUT]
  float* out = (float*)d_out;

  const size_t WELEM = (size_t)NLAY * HID * HID;       // 8,388,608
  const size_t SELEM = (size_t)SEQT * BATCH * HID;     // 8,388,608
  const size_t FELEM = (size_t)FCOUT * HID;            // 131,072

  _Float16* wih16 = (_Float16*)d_ws;
  _Float16* whh16 = wih16 + WELEM;
  _Float16* wfc16 = whh16 + WELEM;
  _Float16* seqA  = wfc16 + FELEM;
  _Float16* seqB  = seqA + SELEM;
  (void)ws_size; (void)in_sizes; (void)n_in; (void)out_size;

  // one-shot f16 conversions + time-major transpose of x
  cvt_f32_to_f16<<<(unsigned)((WELEM + 255) / 256), 256, 0, stream>>>(W_ih, wih16, (int)WELEM);
  cvt_f32_to_f16<<<(unsigned)((WELEM + 255) / 256), 256, 0, stream>>>(W_hh, whh16, (int)WELEM);
  cvt_f32_to_f16<<<(unsigned)((FELEM + 255) / 256), 256, 0, stream>>>(W_fc, wfc16, (int)FELEM);
  transpose_x_kernel<<<(unsigned)((SELEM + 255) / 256), 256, 0, stream>>>(x, seqA);

  // 32 stacked layers, ping-pong the time-major activation sequence
  _Float16* sin  = seqA;
  _Float16* sout = seqB;
  for (int l = 0; l < NLAY; ++l) {
    rnn_layer_kernel<<<1, 1024, 0, stream>>>(
        wih16 + (size_t)l * HID * HID,
        whh16 + (size_t)l * HID * HID,
        b_ih + (size_t)l * HID,
        b_hh + (size_t)l * HID,
        sin, sout);
    _Float16* tmp = sin; sin = sout; sout = tmp;
  }

  // FC head on the last layer's final hidden state (t = T-1)
  fc_kernel<<<1, 512, 0, stream>>>(sin + (size_t)(SEQT - 1) * BATCH * HID,
                                   wfc16, b_fc, out);
}
